// TransformerModels_23991687315664
// MI455X (gfx1250) — matrix operand order repaired and checked by reading, not compiled, not run on hardware
//
#include <hip/hip_runtime.h>
#include <hip/hip_bf16.h>
#include <stdint.h>

// ---------------------------------------------------------------------------
// GPT-style transformer forward for MI455X (gfx1250, wave32, WMMA).
// bf16-native GEMMs: weights pre-converted to bf16, activations produced in
// bf16, tiles DMA'd into LDS with global_load_async_to_lds_b128 (ASYNCcnt),
// double-buffered, consumed by v_wmma_f32_16x16x32_bf16.
// Wide kernel: 64x128x64 block tile, 32x32 wave tile (4 WMMA / 8 ds_load).
// ---------------------------------------------------------------------------

typedef __attribute__((ext_vector_type(16))) __bf16        v16bf;
typedef __attribute__((ext_vector_type(8)))  float         v8f;
typedef __attribute__((ext_vector_type(8)))  unsigned int  v8u;

static constexpr int B_  = 8;
static constexpr int T_  = 512;
static constexpr int V_  = 32000;
static constexpr int D_  = 1024;
static constexpr int H_  = 16;
static constexpr int L_  = 4;
static constexpr int FF_ = 4096;
static constexpr int DK_ = D_ / H_;   // 64
static constexpr int BT_ = B_ * T_;   // 4096
static constexpr int F3_ = 3 * D_;    // 3072

__device__ __forceinline__ unsigned short f2bf(float f) {
  unsigned u = __float_as_uint(f);
  u += 0x7FFFu + ((u >> 16) & 1u);
  return (unsigned short)(u >> 16);
}
__device__ __forceinline__ float bf2f(unsigned short h) {
  return __uint_as_float(((unsigned)h) << 16);
}
// low 32 bits of a flat pointer to __shared__ == LDS byte offset (ISA 10.2)
__device__ __forceinline__ unsigned lds_off(const void* p) {
  return (unsigned)(unsigned long long)p;
}
__device__ __forceinline__ void async_b128(unsigned lds, const void* g) {
  asm volatile("global_load_async_to_lds_b128 %0, %1, off"
               :: "v"(lds), "v"((unsigned long long)g) : "memory");
}

// ---------------------------------------------------------------------------
// fp32 -> bf16 bulk converter (float4 in, packed uint4 out; n % 2048 == 0)
// ---------------------------------------------------------------------------
__global__ __launch_bounds__(256)
void cvt_bf16_kernel(const float* __restrict__ in, unsigned short* __restrict__ out)
{
  long i = ((long)blockIdx.x * 256 + threadIdx.x) * 8;
  float4 a = *(const float4*)(in + i);
  float4 b = *(const float4*)(in + i + 4);
  union { unsigned short us[8]; uint4 u4; } p;
  p.us[0] = f2bf(a.x); p.us[1] = f2bf(a.y); p.us[2] = f2bf(a.z); p.us[3] = f2bf(a.w);
  p.us[4] = f2bf(b.x); p.us[5] = f2bf(b.y); p.us[6] = f2bf(b.z); p.us[7] = f2bf(b.w);
  *(uint4*)(out + i) = p.u4;
}

// ---------------------------------------------------------------------------
// Wide WMMA GEMM (B in [N,K] "weight" layout, async-DMA staged):
//   C[z] = alpha * A[z] (MxK) * B[z]^T + bias (+ C if ACC)
// Block tile 64(M) x 128(N) x 64(K), 256 threads = 8 wave32 in 2(M) x 4(N).
// Each wave owns a 32x32 tile: 2 A frags x 2 B frags -> 4 WMMAs per 32-K.
// Double-buffered LDS; per-tile async ops per wave = 6 (2 A + 4 B).
// Dims must divide the tile (true for all uses here).
// ---------------------------------------------------------------------------
template <bool ACC, bool RELU, bool OUTBF>
__global__ __launch_bounds__(256)
void gemm_wide_wmma(const unsigned short* __restrict__ Ag,
                    const unsigned short* __restrict__ Bg,
                    void* __restrict__ Cv, const float* __restrict__ bias,
                    int K, long lda, long ldb, long ldc,
                    int batch1, long sA0, long sA1, long sB0, long sB1,
                    long sC0, long sC1, float alpha)
{
  constexpr int BM = 64, BN = 128, BK = 64, LDK = BK + 8;
  __shared__ __align__(16) unsigned short As[2][BM][LDK];
  __shared__ __align__(16) unsigned short Bs[2][BN][LDK];

  const int z  = blockIdx.z;
  const int b0 = z / batch1;
  const int b1 = z % batch1;
  Ag += (long)b0 * sA0 + (long)b1 * sA1;
  Bg += (long)b0 * sB0 + (long)b1 * sB1;

  const int tid  = threadIdx.x;
  const int lane = tid & 31;
  const int wid  = tid >> 5;
  const int m0   = (wid & 1) * 32;    // 2 waves along M
  const int n0   = (wid >> 1) * 32;   // 4 waves along N
  const int half = lane >> 4;
  const int lr   = lane & 15;
  const long rowM = (long)blockIdx.y * BM;
  const long colN = (long)blockIdx.x * BN;

  const int sr = tid >> 3;            // staging row base
  const int sc = (tid & 7) * 8;       // staging col (8 bf16 = 16B chunks)

  auto issueA = [&](int kk, int buf) {
#pragma unroll
    for (int j = 0; j < 2; ++j) {     // rows sr, sr+32
      int r = sr + j * 32;
      async_b128(lds_off(&As[buf][r][sc]), Ag + (rowM + r) * lda + kk + sc);
    }
  };
  auto issueB = [&](int kk, int buf) {
#pragma unroll
    for (int j = 0; j < 4; ++j) {     // rows sr, sr+32, sr+64, sr+96
      int r = sr + j * 32;
      async_b128(lds_off(&Bs[buf][r][sc]), Bg + (colN + r) * ldb + kk + sc);
    }
  };

  v8f a00 = {}, a01 = {}, a10 = {}, a11 = {};
  const int nst = K / BK;

  issueA(0, 0);
  issueB(0, 0);

  for (int i = 0; i < nst; ++i) {
    const int cur = i & 1;
    if (i + 1 < nst) {
      issueA((i + 1) * BK, cur ^ 1);
      issueB((i + 1) * BK, cur ^ 1);
      asm volatile("s_wait_asynccnt 6" ::: "memory");  // tile i (oldest 6) done
    } else {
      asm volatile("s_wait_asynccnt 0" ::: "memory");
    }
    __syncthreads();

#pragma unroll
    for (int ks = 0; ks < BK; ks += 32) {
      v8u u0, u1, w0, w1;
#pragma unroll
      for (int v = 0; v < 8; ++v) {
        // A (16x32 bf16) fragment layout, ISA 7.12.2
        int ka = ks + 2 * (v & 3) + ((v >> 2) << 4) + half * 8;
        u0[v] = *(const unsigned int*)&As[cur][m0 + lr][ka];
        u1[v] = *(const unsigned int*)&As[cur][m0 + 16 + lr][ka];
        // B (32x16 bf16) fragment layout
        int kb = ks + 2 * (v & 3) + ((v >> 2) << 4) + half * 8;
        w0[v] = *(const unsigned int*)&Bs[cur][n0 + lr][kb];
        w1[v] = *(const unsigned int*)&Bs[cur][n0 + 16 + lr][kb];
      }
      v16bf fa0 = __builtin_bit_cast(v16bf, u0);
      v16bf fa1 = __builtin_bit_cast(v16bf, u1);
      v16bf fb0 = __builtin_bit_cast(v16bf, w0);
      v16bf fb1 = __builtin_bit_cast(v16bf, w1);
      a00 = __builtin_amdgcn_wmma_f32_16x16x32_bf16(false, fa0, false, fb0, (short)0, a00, false, false);
      a01 = __builtin_amdgcn_wmma_f32_16x16x32_bf16(false, fa0, false, fb1, (short)0, a01, false, false);
      a10 = __builtin_amdgcn_wmma_f32_16x16x32_bf16(false, fa1, false, fb0, (short)0, a10, false, false);
      a11 = __builtin_amdgcn_wmma_f32_16x16x32_bf16(false, fa1, false, fb1, (short)0, a11, false, false);
    }
    __syncthreads();
  }

  // epilogue: C VGPR v -> row base + v + 8*half, col base + lane%16
  float* Cf = (float*)Cv;
  unsigned short* Cb = (unsigned short*)Cv;
  const long cofs = (long)b0 * sC0 + (long)b1 * sC1;
#pragma unroll
  for (int mi = 0; mi < 2; ++mi) {
#pragma unroll
    for (int g = 0; g < 2; ++g) {
      const v8f& acc = mi ? (g ? a11 : a10) : (g ? a01 : a00);
#pragma unroll
      for (int v = 0; v < 8; ++v) {
        long r   = rowM + m0 + mi * 16 + v + 8 * half;
        long cc  = colN + n0 + g * 16 + lr;
        long idx = cofs + r * ldc + cc;
        float val = alpha * acc[v];
        if (bias) val += bias[cc];
        if (ACC)  val += Cf[idx];
        if (RELU) val = fmaxf(val, 0.0f);
        if (OUTBF) Cb[idx] = f2bf(val);
        else       Cf[idx] = val;
      }
    }
  }
}

// ---------------------------------------------------------------------------
// Attention-context GEMM: C = A (MxK, bf16 probs) * B ([K,N] row-major V).
// 64x64x64 tile, 8 waves of 16x32, async A staging, transposed B staging.
// Used only for N = DK = 64.
// ---------------------------------------------------------------------------
__global__ __launch_bounds__(256)
void gemm_kn_wmma(const unsigned short* __restrict__ Ag,
                  const unsigned short* __restrict__ Bg,
                  unsigned short* __restrict__ Cb,
                  int K, long lda, long ldb, long ldc,
                  int batch1, long sA0, long sA1, long sB0, long sB1,
                  long sC0, long sC1)
{
  constexpr int BM = 64, BN = 64, BK = 64, LDK = BK + 8;
  __shared__ __align__(16) unsigned short As[2][BM][LDK];
  __shared__ __align__(16) unsigned short Bs[2][BN][LDK];

  const int z  = blockIdx.z;
  const int b0 = z / batch1;
  const int b1 = z % batch1;
  Ag += (long)b0 * sA0 + (long)b1 * sA1;
  Bg += (long)b0 * sB0 + (long)b1 * sB1;

  const int tid  = threadIdx.x;
  const int lane = tid & 31;
  const int wid  = tid >> 5;
  const int m0   = (wid & 3) * 16;
  const int n0   = (wid >> 2) * 32;
  const int half = lane >> 4;
  const int lr   = lane & 15;
  const long rowM = (long)blockIdx.y * BM;
  const long colN = (long)blockIdx.x * BN;

  const int sr = tid >> 3;
  const int sc = (tid & 7) * 8;

  auto issueA = [&](int kk, int buf) {
#pragma unroll
    for (int j = 0; j < 2; ++j) {
      int r = sr + j * 32;
      async_b128(lds_off(&As[buf][r][sc]), Ag + (rowM + r) * lda + kk + sc);
    }
  };
  // transposed staging: vector global read along n, b16 LDS writes Bs[n][k]
  auto issueB = [&](int kk, int buf) {
#pragma unroll
    for (int j = 0; j < 2; ++j) {
      int ch = tid + j * 256;
      int kI = ch >> 3;
      int nI = (ch & 7) * 8;
      uint4 raw = *(const uint4*)(Bg + (long)(kk + kI) * ldb + colN + nI);
      union { uint4 u4; unsigned short us[8]; } u; u.u4 = raw;
#pragma unroll
      for (int q = 0; q < 8; ++q) Bs[buf][nI + q][kI] = u.us[q];
    }
  };

  v8f acc0 = {}, acc1 = {};
  const int nst = K / BK;

  issueA(0, 0);
  issueB(0, 0);

  for (int i = 0; i < nst; ++i) {
    const int cur = i & 1;
    if (i + 1 < nst) {
      issueA((i + 1) * BK, cur ^ 1);
      issueB((i + 1) * BK, cur ^ 1);
      asm volatile("s_wait_asynccnt 2" ::: "memory");
    } else {
      asm volatile("s_wait_asynccnt 0" ::: "memory");
    }
    __syncthreads();

#pragma unroll
    for (int ks = 0; ks < BK; ks += 32) {
      v8u au, bu0, bu1;
#pragma unroll
      for (int v = 0; v < 8; ++v) {
        int ka = ks + 2 * (v & 3) + ((v >> 2) << 4) + half * 8;
        au[v] = *(const unsigned int*)&As[cur][m0 + lr][ka];
        int kb = ks + 2 * (v & 3) + ((v >> 2) << 4) + half * 8;
        bu0[v] = *(const unsigned int*)&Bs[cur][n0 + lr][kb];
        bu1[v] = *(const unsigned int*)&Bs[cur][n0 + 16 + lr][kb];
      }
      v16bf af  = __builtin_bit_cast(v16bf, au);
      v16bf fb0 = __builtin_bit_cast(v16bf, bu0);
      v16bf fb1 = __builtin_bit_cast(v16bf, bu1);
      acc0 = __builtin_amdgcn_wmma_f32_16x16x32_bf16(false, af, false, fb0, (short)0, acc0, false, false);
      acc1 = __builtin_amdgcn_wmma_f32_16x16x32_bf16(false, af, false, fb1, (short)0, acc1, false, false);
    }
    __syncthreads();
  }

  const long cofs = (long)b0 * sC0 + (long)b1 * sC1;
#pragma unroll
  for (int v = 0; v < 8; ++v) {
    long r = rowM + m0 + v + 8 * half;
    Cb[cofs + r * ldc + colN + n0 + lr]      = f2bf(acc0[v]);
    Cb[cofs + r * ldc + colN + n0 + 16 + lr] = f2bf(acc1[v]);
  }
}

// ---------------------------------------------------------------------------
// Embedding gather (fp32 residual stream): hidden[row,:] = emb[x[row],:]
// ---------------------------------------------------------------------------
__global__ __launch_bounds__(256)
void embed_kernel(const int* __restrict__ x, const float* __restrict__ emb,
                  float* __restrict__ hidden)
{
  long row = blockIdx.x;
  long tok = x[row];
  const float4* src = (const float4*)(emb + tok * D_);
  float4* dst = (float4*)(hidden + row * D_);
  dst[threadIdx.x] = src[threadIdx.x];
}

// ---------------------------------------------------------------------------
// LayerNorm over D=1024: fp32 in, bf16 out. One 256-thread block per row.
// ---------------------------------------------------------------------------
__global__ __launch_bounds__(256)
void layernorm_kernel(const float* __restrict__ X, const float* __restrict__ w,
                      const float* __restrict__ b, unsigned short* __restrict__ Y)
{
  __shared__ float rsum[8], rsq[8];
  long row = blockIdx.x;
  const float4 v = ((const float4*)(X + row * D_))[threadIdx.x];

  float s  = v.x + v.y + v.z + v.w;
  float sq = v.x * v.x + v.y * v.y + v.z * v.z + v.w * v.w;
#pragma unroll
  for (int o = 16; o; o >>= 1) {
    s  += __shfl_xor(s, o, 32);
    sq += __shfl_xor(sq, o, 32);
  }
  int wid = threadIdx.x >> 5;
  if ((threadIdx.x & 31) == 0) { rsum[wid] = s; rsq[wid] = sq; }
  __syncthreads();
  s = 0.0f; sq = 0.0f;
#pragma unroll
  for (int i = 0; i < 8; ++i) { s += rsum[i]; sq += rsq[i]; }
  float mu  = s * (1.0f / D_);
  float var = sq * (1.0f / D_) - mu * mu;
  float rs  = rsqrtf(var + 1e-5f);

  const float4 wv = ((const float4*)w)[threadIdx.x];
  const float4 bv = ((const float4*)b)[threadIdx.x];
  unsigned short o0 = f2bf((v.x - mu) * rs * wv.x + bv.x);
  unsigned short o1 = f2bf((v.y - mu) * rs * wv.y + bv.y);
  unsigned short o2 = f2bf((v.z - mu) * rs * wv.z + bv.z);
  unsigned short o3 = f2bf((v.w - mu) * rs * wv.w + bv.w);
  uint2 pk;
  pk.x = (unsigned)o0 | ((unsigned)o1 << 16);
  pk.y = (unsigned)o2 | ((unsigned)o3 << 16);
  ((uint2*)(Y + row * D_))[threadIdx.x] = pk;
}

// ---------------------------------------------------------------------------
// RoPE in place on the bf16 qkv buffer (per token: H x [q64 | k64 | v64]).
// ---------------------------------------------------------------------------
__global__ __launch_bounds__(256)
void rope_kernel(unsigned short* __restrict__ qkv)
{
  int idx = blockIdx.x * 256 + threadIdx.x;   // B*T*H*(DK/2) = 2^21
  int p = idx & 31;
  int h = (idx >> 5) & (H_ - 1);
  int t = (idx >> 9) & (T_ - 1);
  int b = idx >> 18;
  long base = (long)(b * T_ + t) * F3_ + h * (3 * DK_);
  float ang = (float)t * __powf(10000.0f, -(float)(2 * p) / (float)DK_);
  float sn, cs;
  __sincosf(ang, &sn, &cs);

  float q0 = bf2f(qkv[base + 2 * p]), q1 = bf2f(qkv[base + 2 * p + 1]);
  qkv[base + 2 * p]     = f2bf(q0 * cs - q1 * sn);
  qkv[base + 2 * p + 1] = f2bf(q0 * sn + q1 * cs);

  float k0 = bf2f(qkv[base + DK_ + 2 * p]), k1 = bf2f(qkv[base + DK_ + 2 * p + 1]);
  qkv[base + DK_ + 2 * p]     = f2bf(k0 * cs - k1 * sn);
  qkv[base + DK_ + 2 * p + 1] = f2bf(k0 * sn + k1 * cs);
}

// ---------------------------------------------------------------------------
// Causal softmax in place on bf16 scores (fp32 math in registers).
// One wave32 per row of T=512; 8 rows per block.
// ---------------------------------------------------------------------------
__global__ __launch_bounds__(256)
void softmax_causal(unsigned short* __restrict__ S)
{
  int wid  = threadIdx.x >> 5;
  int lane = threadIdx.x & 31;
  long row = (long)blockIdx.x * 8 + wid;   // B*H*T rows
  int t = (int)(row & (T_ - 1));
  unsigned short* p = S + row * T_;

  float x[16];
  float mx = -__builtin_inff();
#pragma unroll
  for (int i = 0; i < 16; ++i) {
    int c = lane + i * 32;
    float v = (c <= t) ? bf2f(p[c]) : -__builtin_inff();
    x[i] = v;
    mx = fmaxf(mx, v);
  }
#pragma unroll
  for (int o = 16; o; o >>= 1) mx = fmaxf(mx, __shfl_xor(mx, o, 32));

  float s = 0.0f;
#pragma unroll
  for (int i = 0; i < 16; ++i) {
    int c = lane + i * 32;
    float e = (c <= t) ? __expf(x[i] - mx) : 0.0f;
    x[i] = e;
    s += e;
  }
#pragma unroll
  for (int o = 16; o; o >>= 1) s += __shfl_xor(s, o, 32);
  float inv = 1.0f / s;
#pragma unroll
  for (int i = 0; i < 16; ++i) p[lane + i * 32] = f2bf(x[i] * inv);
}

// ---------------------------------------------------------------------------
// Host-side orchestration.
// ---------------------------------------------------------------------------
extern "C" void kernel_launch(void* const* d_in, const int* in_sizes, int n_in,
                              void* d_out, int out_size, void* d_ws, size_t ws_size,
                              hipStream_t stream)
{
  (void)in_sizes; (void)n_in; (void)out_size; (void)ws_size;

  const int*   x         = (const int*)d_in[0];
  const float* token_emb = (const float*)d_in[1];
  const float* ln1_w     = (const float*)d_in[2];
  const float* ln1_b     = (const float*)d_in[3];
  const float* qkv_w     = (const float*)d_in[4];
  const float* qkv_b     = (const float*)d_in[5];
  const float* out_w     = (const float*)d_in[6];
  const float* out_b     = (const float*)d_in[7];
  const float* ln2_w     = (const float*)d_in[8];
  const float* ln2_b     = (const float*)d_in[9];
  const float* ff1_w     = (const float*)d_in[10];
  const float* ff1_b     = (const float*)d_in[11];
  const float* ff2_w     = (const float*)d_in[12];
  const float* ff2_b     = (const float*)d_in[13];
  const float* lnf_w     = (const float*)d_in[14];
  const float* lnf_b     = (const float*)d_in[15];
  const float* head_w    = (const float*)d_in[16];
  const float* head_b    = (const float*)d_in[17];
  float* out = (float*)d_out;

  // ---- workspace carve-up ----
  char* wp = (char*)d_ws;
  auto take = [&](size_t bytes) {
    char* p = wp;
    wp += (bytes + 255) & ~(size_t)255;
    return (void*)p;
  };
  float*          hidden = (float*)         take((size_t)BT_ * D_ * 4);
  unsigned short* nbuf   = (unsigned short*)take((size_t)BT_ * D_ * 2);
  unsigned short* qkvb   = (unsigned short*)take((size_t)BT_ * F3_ * 2);
  unsigned short* scores = (unsigned short*)take((size_t)B_ * H_ * T_ * T_ * 2);
  unsigned short* actx   = (unsigned short*)take((size_t)BT_ * D_ * 2);
  unsigned short* ffb    = (unsigned short*)take((size_t)BT_ * FF_ * 2);
  unsigned short* wqkv   = (unsigned short*)take((size_t)L_ * F3_ * D_ * 2);
  unsigned short* wout   = (unsigned short*)take((size_t)L_ * D_ * D_ * 2);
  unsigned short* wff1   = (unsigned short*)take((size_t)L_ * FF_ * D_ * 2);
  unsigned short* wff2   = (unsigned short*)take((size_t)L_ * D_ * FF_ * 2);
  unsigned short* whead  = (unsigned short*)take((size_t)V_ * D_ * 2);

  dim3 blk(256);
  const float scale = 0.125f;   // 1/sqrt(DK)

  // ---- weight conversion to bf16 ----
  auto cvt = [&](const float* src, unsigned short* dst, long n) {
    cvt_bf16_kernel<<<dim3((unsigned)(n / 2048)), blk, 0, stream>>>(src, dst);
  };
  cvt(qkv_w,  wqkv,  (long)L_ * F3_ * D_);
  cvt(out_w,  wout,  (long)L_ * D_ * D_);
  cvt(ff1_w,  wff1,  (long)L_ * FF_ * D_);
  cvt(ff2_w,  wff2,  (long)L_ * D_ * FF_);
  cvt(head_w, whead, (long)V_ * D_);

  embed_kernel<<<BT_, blk, 0, stream>>>(x, token_emb, hidden);

  for (int l = 0; l < L_; ++l) {
    layernorm_kernel<<<BT_, blk, 0, stream>>>(hidden, ln1_w + l * D_, ln1_b + l * D_, nbuf);

    // QKV projection -> bf16 qkv buffer
    gemm_wide_wmma<false, false, true><<<dim3(F3_ / 128, BT_ / 64, 1), blk, 0, stream>>>(
        nbuf, wqkv + (size_t)l * F3_ * D_, qkvb, qkv_b + (size_t)l * F3_,
        D_, D_, D_, F3_, 1, 0, 0, 0, 0, 0, 0, 1.0f);

    rope_kernel<<<(B_ * T_ * H_ * (DK_ / 2)) / 256, blk, 0, stream>>>(qkvb);

    // scores = scale * Q K^T  (bf16 out), batched over (b,h)
    gemm_wide_wmma<false, false, true><<<dim3(T_ / 128, T_ / 64, B_ * H_), blk, 0, stream>>>(
        qkvb, qkvb + DK_, scores, nullptr,
        DK_, F3_, F3_, T_, H_,
        (long)T_ * F3_, (long)(3 * DK_),
        (long)T_ * F3_, (long)(3 * DK_),
        (long)H_ * T_ * T_, (long)T_ * T_,
        scale);

    softmax_causal<<<(B_ * H_ * T_) / 8, blk, 0, stream>>>(scores);

    // context = P V  (bf16 out), V is [K=T, N=DK] row-major
    gemm_kn_wmma<<<dim3(DK_ / 64, T_ / 64, B_ * H_), blk, 0, stream>>>(
        scores, qkvb + 2 * DK_, actx,
        T_, T_, F3_, D_, H_,
        (long)H_ * T_ * T_, (long)T_ * T_,
        (long)T_ * F3_, (long)(3 * DK_),
        (long)T_ * D_, (long)DK_);

    // out projection accumulated into fp32 residual stream
    gemm_wide_wmma<true, false, false><<<dim3(D_ / 128, BT_ / 64, 1), blk, 0, stream>>>(
        actx, wout + (size_t)l * D_ * D_, hidden, out_b + (size_t)l * D_,
        D_, D_, D_, D_, 1, 0, 0, 0, 0, 0, 0, 1.0f);

    layernorm_kernel<<<BT_, blk, 0, stream>>>(hidden, ln2_w + l * D_, ln2_b + l * D_, nbuf);

    // FF1 + ReLU -> bf16
    gemm_wide_wmma<false, true, true><<<dim3(FF_ / 128, BT_ / 64, 1), blk, 0, stream>>>(
        nbuf, wff1 + (size_t)l * FF_ * D_, ffb, ff1_b + (size_t)l * FF_,
        D_, D_, D_, FF_, 1, 0, 0, 0, 0, 0, 0, 1.0f);

    // FF2 accumulated into fp32 residual stream
    gemm_wide_wmma<true, false, false><<<dim3(D_ / 128, BT_ / 64, 1), blk, 0, stream>>>(
        ffb, wff2 + (size_t)l * D_ * FF_, hidden, ff2_b + (size_t)l * D_,
        FF_, FF_, FF_, D_, 1, 0, 0, 0, 0, 0, 0, 1.0f);
  }

  // final LN + vocab head -> fp32 logits in d_out [4096, 32000]
  layernorm_kernel<<<BT_, blk, 0, stream>>>(hidden, lnf_w, lnf_b, nbuf);
  gemm_wide_wmma<false, false, false><<<dim3(V_ / 128, BT_ / 64, 1), blk, 0, stream>>>(
      nbuf, whead, out, head_b,
      D_, D_, D_, V_, 1, 0, 0, 0, 0, 0, 0, 1.0f);
}
